// Generator_21139829031546
// MI455X (gfx1250) — matrix operand order repaired and checked by reading, not compiled, not run on hardware
//
#include <hip/hip_runtime.h>

#define E_EDGES   500000
#define NEG_SLOPE 0.2f
#define BIG_NEGF  -1e30f

typedef __attribute__((ext_vector_type(16))) _Float16 v16h;
typedef __attribute__((ext_vector_type(8)))  _Float16 v8h;
typedef __attribute__((ext_vector_type(8)))  float    v8f;

static __device__ __forceinline__ float lrelu(float v) { return v > 0.f ? v : NEG_SLOPE * v; }

// float atomic max via int/uint ordering trick (works for mixed signs)
static __device__ __forceinline__ void atomicMaxF(float* addr, float v) {
  if (v >= 0.f) atomicMax((int*)addr, __float_as_int(v));
  else          atomicMin((unsigned int*)addr, __float_as_uint(v));
}

// ---------------------------------------------------------------- fills
__global__ void k_fill_f32(float* p, float v, long n) {
  long i = (long)blockIdx.x * blockDim.x + threadIdx.x;
  if (i < n) p[i] = v;
}
__global__ void k_fill_i32(int* p, int v, long n) {
  long i = (long)blockIdx.x * blockDim.x + threadIdx.x;
  if (i < n) p[i] = v;
}
// out[n, C] = bias[C] broadcast
__global__ void k_bias_rows(float* out, const float* b, long n, int C) {
  long i = (long)blockIdx.x * blockDim.x + threadIdx.x;
  if (i < n * C) out[i] = b[i % C];
}

// ---------------------------------------------------------------- converts
__global__ void k_f32_to_f16(const float* s, _Float16* d, long n) {
  long i = (long)blockIdx.x * blockDim.x + threadIdx.x;
  if (i < n) d[i] = (_Float16)s[i];
}
// W[K,N] f32 -> Wt[N,K] f16 (transpose so B-fragment loads are contiguous)
__global__ void k_w_to_f16t(const float* W, _Float16* Wt, int K, int N) {
  long i = (long)blockIdx.x * blockDim.x + threadIdx.x;
  if (i >= (long)K * N) return;
  int k = (int)(i / N), nn = (int)(i % N);
  Wt[(size_t)nn * K + k] = (_Float16)W[i];
}
// decoder concat: out[n, ac+bc] = [ a[row % arows, :ac] | b[row, :bc] ]  (f32 -> f16)
__global__ void k_concat_f16(const float* a, int arows, int ac,
                             const float* b, int bc, _Float16* o, int n) {
  int C = ac + bc;
  long i = (long)blockIdx.x * blockDim.x + threadIdx.x;
  if (i >= (long)n * C) return;
  int row = (int)(i / C), col = (int)(i % C);
  float v = (col < ac) ? a[(size_t)(row % arows) * ac + col]
                       : b[(size_t)row * bc + (col - ac)];
  o[i] = (_Float16)v;
}

// ---------------------------------------------------------------- WMMA GEMM
// C[M,N] = A[M,K](f16, row-major) x Bt[N,K](f16, B transposed) ; fp32 accumulate.
// - B tile [16,K] is staged once per block into LDS via async global->LDS copy
//   (global_load_async_to_lds_b128 / ASYNCcnt / s_wait_asynccnt): all 4 waves in
//   the block share tn, so this cuts global B traffic 4x.
// - K-loop is register double-buffered so the s_wait before each WMMA covers
//   only loads issued one full iteration earlier.
// A-frag layout: lanes<16 K=k..k+15, lanes>=16 K=k+16..k+31 (matches B-frag).
// B-frag: lane holds column N=lane%16; lanes<16 K=k..k+15, lanes>=16 K=k+16..k+31.
#define GEMM_WAVES 4
#define MAX_K 320
__global__ void __launch_bounds__(32 * GEMM_WAVES)
k_gemm_wmma(const _Float16* __restrict__ A, const _Float16* __restrict__ Bt,
            float* __restrict__ C, int M, int N, int K) {
  __shared__ __align__(32) _Float16 btile[16 * MAX_K];
  int tn = blockIdx.y;

  // ---- async stage of the shared B tile (all threads participate, then barrier)
  {
    const _Float16* gsrc = Bt + (size_t)tn * 16 * K;       // contiguous 16*K halfs
    unsigned lbase = (unsigned)(uintptr_t)(&btile[0]);     // LDS byte offset
    int chunks = 2 * K;                                    // (16*K halfs) / 8 per 16B
    for (int t = threadIdx.x; t < chunks; t += blockDim.x) {
      const _Float16* g = gsrc + (size_t)t * 8;
      unsigned l = lbase + (unsigned)t * 16u;
      asm volatile("global_load_async_to_lds_b128 %0, %1, off"
                   :: "v"(l), "v"(g) : "memory");
    }
    asm volatile("s_wait_asynccnt 0x0" ::: "memory");      // drain this wave's ASYNCcnt
    __syncthreads();                                       // tile visible block-wide
  }

  int lane = threadIdx.x & 31;
  int wave = threadIdx.x >> 5;
  int tm = blockIdx.x * GEMM_WAVES + wave;
  if (tm * 16 >= M) return;
  int r     = lane & 15;
  int abase = (lane < 16) ? 0 : 16;
  int bbase = (lane < 16) ? 0 : 16;
  const _Float16* arow = A + (size_t)(tm * 16 + r) * K;
  const _Float16* brow = &btile[r * K];

  union AB { v16h v; v8h h[2]; };
  v8f acc = {};
  AB a_cur, a_nxt = {};
  v16h b_cur, b_nxt = {};
  a_cur.h[0] = *(const v8h*)(arow + abase);
  a_cur.h[1] = *(const v8h*)(arow + abase + 8);
  b_cur = *(const v16h*)(brow + bbase);
  for (int k = 0; k < K; k += 32) {
    int kn = k + 32;
    if (kn < K) {                                          // issue next-iter fragments
      __builtin_prefetch(arow + kn + 32, 0, 0);            // global_prefetch_b8
      a_nxt.h[0] = *(const v8h*)(arow + kn + abase);
      a_nxt.h[1] = *(const v8h*)(arow + kn + abase + 8);
      b_nxt = *(const v16h*)(brow + kn + bbase);
    }
    acc = __builtin_amdgcn_wmma_f32_16x16x32_f16(false, a_cur.v, false, b_cur,
                                                 (short)0, acc, false, false);
    a_cur = a_nxt;
    b_cur = b_nxt;
  }
  int col     = tn * 16 + r;
  int rowbase = tm * 16 + ((lane < 16) ? 0 : 8);
#pragma unroll
  for (int i = 0; i < 8; ++i)
    C[(size_t)(rowbase + i) * N + col] = acc[i];
}

// ---------------------------------------------------------------- attention
// as[n,h] = <h[n,h,:], a_src[h,:]>, ad likewise
__global__ void k_attn_scores(const float* __restrict__ h, const float* __restrict__ asrc,
                              const float* __restrict__ adst, float* as_, float* ad_,
                              int n, int H, int D) {
  int i = blockIdx.x * blockDim.x + threadIdx.x;
  if (i >= n * H) return;
  int nn = i / H, hh = i % H;
  const float* hp = h + (size_t)nn * H * D + (size_t)hh * D;
  const float* ap = asrc + (size_t)hh * D;
  const float* bp = adst + (size_t)hh * D;
  float s = 0.f, t = 0.f;
  for (int d = 0; d < D; ++d) { float v = hp[d]; s += v * ap[d]; t += v * bp[d]; }
  as_[i] = s; ad_[i] = t;
}

// segment max over dst (init BIG_NEG)
__global__ void k_edge_max(const int* src, const int* dst, const int* mask,
                           const float* as_, const float* ad_, float* m, int H) {
  int e = blockIdx.x * blockDim.x + threadIdx.x;
  if (e >= E_EDGES || !mask[e]) return;
  int s = src[e], d = dst[e];
  for (int h = 0; h < H; ++h) {
    float v = lrelu(as_[s * H + h] + ad_[d * H + h]);
    atomicMaxF(&m[d * H + h], v);
  }
}

// ex = exp(e - m[dst]) * mask ; den = segment_sum(ex)
__global__ void k_edge_exp(const int* src, const int* dst, const int* mask,
                           const float* as_, const float* ad_, const float* m,
                           float* den, float* exbuf, int H) {
  int e = blockIdx.x * blockDim.x + threadIdx.x;
  if (e >= E_EDGES) return;
  int ok = mask[e];
  int s = src[e], d = dst[e];
  for (int h = 0; h < H; ++h) {
    float exv = 0.f;
    if (ok) {
      float v = lrelu(as_[s * H + h] + ad_[d * H + h]);
      exv = __expf(v - m[d * H + h]);
      atomicAdd(&den[d * H + h], exv);
    }
    exbuf[(size_t)e * H + h] = exv;
  }
}

// out[dst, c] += h[src, c] * ex/den  (out pre-filled with bias)
__global__ void k_aggregate(const int* __restrict__ src, const int* __restrict__ dst,
                            const int* __restrict__ mask, const float* __restrict__ hfeat,
                            const float* __restrict__ exbuf, const float* __restrict__ den,
                            float* __restrict__ out, int H, int D) {
  int C = H * D;
  long i = (long)blockIdx.x * blockDim.x + threadIdx.x;
  if (i >= (long)E_EDGES * C) return;
  int e = (int)(i / C), c = (int)(i % C);
  if (!mask[e]) return;
  int hh = c / D;
  float ex = exbuf[(size_t)e * H + hh];
  if (ex == 0.f) return;
  float al = ex / fmaxf(den[(size_t)dst[e] * H + hh], 1e-16f);
  atomicAdd(&out[(size_t)dst[e] * C + c], hfeat[(size_t)src[e] * C + c] * al);
}

// ---------------------------------------------------------------- batchnorm
// LDS-staged per-channel sum / sumsq (ds_add_f32), then one global add per block
__global__ void k_bn_stats(const float* __restrict__ h, float* __restrict__ sums,
                           long n, int C) {
  extern __shared__ float lds[];                  // 2*C floats
  for (int i = threadIdx.x; i < 2 * C; i += blockDim.x) lds[i] = 0.f;
  __syncthreads();
  long total = n * C;
  for (long i = (long)blockIdx.x * blockDim.x + threadIdx.x; i < total;
       i += (long)gridDim.x * blockDim.x) {
    int c = (int)(i % C);
    float v = h[i];
    atomicAdd(&lds[c], v);
    atomicAdd(&lds[C + c], v * v);
  }
  __syncthreads();
  for (int i = threadIdx.x; i < 2 * C; i += blockDim.x) atomicAdd(&sums[i], lds[i]);
}

__global__ void k_bn_apply(float* h, const float* sums, const float* gamma,
                           const float* beta, long n, int C) {
  long i = (long)blockIdx.x * blockDim.x + threadIdx.x;
  if (i >= n * C) return;
  int c = (int)(i % C);
  float invN = 1.f / (float)n;
  float mean = sums[c] * invN;
  float var  = sums[C + c] * invN - mean * mean;
  float y = (h[i] - mean) * rsqrtf(var + 1e-5f) * gamma[c] + beta[c];
  h[i] = lrelu(y);
}

// ---------------------------------------------------------------- top-k pool
__global__ void k_score(const float* __restrict__ h, const float* __restrict__ w,
                        float* score, int n, int C) {
  int i = blockIdx.x * blockDim.x + threadIdx.x;
  if (i >= n) return;
  float dot = 0.f, nrm = 0.f;
  for (int c = 0; c < C; ++c) { float wv = w[c]; dot += h[(size_t)i * C + c] * wv; nrm += wv * wv; }
  score[i] = tanhf(dot * rsqrtf(fmaxf(nrm, 1e-30f)));
}

// device-side bisection for the k-th largest (scores in (-1,1)); 1 block
__global__ void k_bisect(const float* __restrict__ score, int n, int k, float* thr) {
  __shared__ int cnt;
  float lo = -1.0001f, hi = 1.0001f;
  for (int it = 0; it < 40; ++it) {
    if (threadIdx.x == 0) cnt = 0;
    __syncthreads();
    float mid = 0.5f * (lo + hi);
    int c = 0;
    for (int i = threadIdx.x; i < n; i += blockDim.x) c += (score[i] > mid) ? 1 : 0;
    atomicAdd(&cnt, c);
    __syncthreads();
    if (cnt > k) lo = mid; else hi = mid;
    __syncthreads();
  }
  if (threadIdx.x == 0) *thr = hi;    // count(score > hi) <= k
}

// two-pass compaction: pass 1 takes score > thr, pass 2 fills remainder
__global__ void k_compact(const float* score, const float* thr, int n, int k,
                          int gtpass, int* counter, int* perm, int* inv) {
  int i = blockIdx.x * blockDim.x + threadIdx.x;
  if (i >= n) return;
  float t = *thr;
  bool sel = gtpass ? (score[i] > t) : !(score[i] > t);
  if (!sel) return;
  int pos = atomicAdd(counter, 1);
  if (pos < k) { perm[pos] = i; inv[i] = pos; }
}

// xp[j,:] = h[perm[j],:] * score[perm[j]]
__global__ void k_gather_scale(const float* h, const int* perm, const float* score,
                               float* xp, int k, int C) {
  long i = (long)blockIdx.x * blockDim.x + threadIdx.x;
  if (i >= (long)k * C) return;
  int j = (int)(i / C), c = (int)(i % C);
  int node = perm[j];
  xp[i] = h[(size_t)node * C + c] * score[node];
}

__global__ void k_relabel(const int* src, const int* dst, const int* mask, const int* inv,
                          int* nsrc, int* ndst, int* nmask) {
  int e = blockIdx.x * blockDim.x + threadIdx.x;
  if (e >= E_EDGES) return;
  int s = inv[src[e]], d = inv[dst[e]];
  int ok = mask[e] && s >= 0 && d >= 0;
  nsrc[e]  = ok ? s : 0;
  ndst[e]  = ok ? d : 0;
  nmask[e] = ok;
}

// ================================================================ host side
static inline long cdiv(long a, long b) { return (a + b - 1) / b; }

extern "C" void kernel_launch(void* const* d_in, const int* in_sizes, int n_in,
                              void* d_out, int out_size, void* d_ws, size_t ws_size,
                              hipStream_t stream) {
  (void)in_sizes; (void)n_in; (void)out_size; (void)ws_size;
  const int TB = 256;
  const int E = E_EDGES;

  // ---- inputs (JAX pytree alphabetical flatten: bot, dec, enc, out, pool)
  const float* x    = (const float*)d_in[0];     // [50000, 64]
  const int*   src0 = (const int*)d_in[1];       // edge_index[0]
  const int*   dst0 = src0 + E;                  // edge_index[1]
  struct BlockP { const float *W, *a_dst, *a_src, *b, *beta, *gamma; };
  auto blk = [&](int base) {
    BlockP p;
    p.W = (const float*)d_in[base];     p.a_dst = (const float*)d_in[base + 1];
    p.a_src = (const float*)d_in[base + 2]; p.b = (const float*)d_in[base + 3];
    p.beta = (const float*)d_in[base + 4];  p.gamma = (const float*)d_in[base + 5];
    return p;
  };
  BlockP BOT = blk(2);
  BlockP DEC0 = blk(8), DEC1 = blk(14), DEC2 = blk(20);
  BlockP ENC0 = blk(26), ENC1 = blk(32), ENC2 = blk(38);
  const float* OUT_W   = (const float*)d_in[44];
  const float* OUT_AD  = (const float*)d_in[45];
  const float* OUT_AS  = (const float*)d_in[46];
  const float* OUT_B   = (const float*)d_in[47];
  const float* POOL0 = (const float*)d_in[48];
  const float* POOL1 = (const float*)d_in[49];
  const float* POOL2 = (const float*)d_in[50];

  // ---- workspace bump allocator
  char* wp = (char*)d_ws;
  auto alloc = [&](size_t bytes) { bytes = (bytes + 255) & ~(size_t)255; void* r = wp; wp += bytes; return r; };
  _Float16* A16   = (_Float16*)alloc((size_t)50000 * 320 * 2);
  _Float16* Wt16  = (_Float16*)alloc((size_t)65536 * 2);
  float* hbuf     = (float*)alloc((size_t)3200000 * 4);
  float* blockout = (float*)alloc((size_t)3200000 * 4);
  float* xp1      = (float*)alloc((size_t)40000 * 32 * 4);
  float* xp2      = (float*)alloc((size_t)24000 * 64 * 4);
  float* xp3      = (float*)alloc((size_t)9600 * 128 * 4);
  float* as_      = (float*)alloc((size_t)200000 * 4);
  float* ad_      = (float*)alloc((size_t)200000 * 4);
  float* m_       = (float*)alloc((size_t)200000 * 4);
  float* den_     = (float*)alloc((size_t)200000 * 4);
  float* exbuf    = (float*)alloc((size_t)2000000 * 4);
  float* sums     = (float*)alloc((size_t)512 * 4);
  float* score    = (float*)alloc((size_t)50000 * 4);
  float* thr      = (float*)alloc(64);
  int*   counter  = (int*)alloc(64);
  int*   inv      = (int*)alloc((size_t)50000 * 4);
  int*   perm     = (int*)alloc((size_t)50000 * 4);
  int*   mask0    = (int*)alloc((size_t)E * 4);
  int *s1 = (int*)alloc((size_t)E * 4), *d1 = (int*)alloc((size_t)E * 4), *m1 = (int*)alloc((size_t)E * 4);
  int *s2 = (int*)alloc((size_t)E * 4), *d2 = (int*)alloc((size_t)E * 4), *m2 = (int*)alloc((size_t)E * 4);
  int *s3 = (int*)alloc((size_t)E * 4), *d3 = (int*)alloc((size_t)E * 4), *m3 = (int*)alloc((size_t)E * 4);

  k_fill_i32<<<cdiv(E, TB), TB, 0, stream>>>(mask0, 1, E);

  // ---- GATConv: h = x@W ; edge softmax over dst ; scatter-add (+bias)
  auto gat_conv = [&](const float* feat /* null => A16 prepared */, int n, int c_in,
                      int c_out, int H, const float* W, const float* aS, const float* aD,
                      const float* b, const int* src, const int* dst, const int* mask,
                      float* out) {
    int D = c_out / H;
    if (feat)
      k_f32_to_f16<<<cdiv((long)n * c_in, TB), TB, 0, stream>>>(feat, A16, (long)n * c_in);
    k_w_to_f16t<<<cdiv((long)c_in * c_out, TB), TB, 0, stream>>>(W, Wt16, c_in, c_out);
    dim3 gg((unsigned)cdiv(n / 16, GEMM_WAVES), (unsigned)(c_out / 16));
    k_gemm_wmma<<<gg, 32 * GEMM_WAVES, 0, stream>>>(A16, Wt16, hbuf, n, c_out, c_in);
    k_attn_scores<<<cdiv((long)n * H, TB), TB, 0, stream>>>(hbuf, aS, aD, as_, ad_, n, H, D);
    k_fill_f32<<<cdiv((long)n * H, TB), TB, 0, stream>>>(m_, BIG_NEGF, (long)n * H);
    k_fill_f32<<<cdiv((long)n * H, TB), TB, 0, stream>>>(den_, 0.f, (long)n * H);
    k_edge_max<<<cdiv(E, TB), TB, 0, stream>>>(src, dst, mask, as_, ad_, m_, H);
    k_edge_exp<<<cdiv(E, TB), TB, 0, stream>>>(src, dst, mask, as_, ad_, m_, den_, exbuf, H);
    k_bias_rows<<<cdiv((long)n * c_out, TB), TB, 0, stream>>>(out, b, n, c_out);
    k_aggregate<<<cdiv((long)E * c_out, TB), TB, 0, stream>>>(src, dst, mask, hbuf, exbuf,
                                                              den_, out, H, D);
  };

  auto gat_block = [&](const float* feat, int n, int c_in, int c_out, const BlockP& P,
                       const int* src, const int* dst, const int* mask, float* out) {
    gat_conv(feat, n, c_in, c_out, 4, P.W, P.a_src, P.a_dst, P.b, src, dst, mask, out);
    k_fill_f32<<<cdiv(2L * c_out, TB), TB, 0, stream>>>(sums, 0.f, 2L * c_out);
    k_bn_stats<<<256, TB, 2 * c_out * (int)sizeof(float), stream>>>(out, sums, n, c_out);
    k_bn_apply<<<cdiv((long)n * c_out, TB), TB, 0, stream>>>(out, sums, P.gamma, P.beta, n, c_out);
  };

  auto pool = [&](const float* h, int n, int k, const float* w, int C,
                  const int* src, const int* dst, const int* mask,
                  float* xp, int* nsrc, int* ndst, int* nmask) {
    k_score<<<cdiv(n, TB), TB, 0, stream>>>(h, w, score, n, C);
    k_bisect<<<1, TB, 0, stream>>>(score, n, k, thr);
    k_fill_i32<<<cdiv(n, TB), TB, 0, stream>>>(inv, -1, n);
    k_fill_i32<<<1, 1, 0, stream>>>(counter, 0, 1);
    k_compact<<<cdiv(n, TB), TB, 0, stream>>>(score, thr, n, k, 1, counter, perm, inv);
    k_compact<<<cdiv(n, TB), TB, 0, stream>>>(score, thr, n, k, 0, counter, perm, inv);
    k_gather_scale<<<cdiv((long)k * C, TB), TB, 0, stream>>>(h, perm, score, xp, k, C);
    k_relabel<<<cdiv(E, TB), TB, 0, stream>>>(src, dst, mask, inv, nsrc, ndst, nmask);
  };

  // ---------------- encoder
  gat_block(x,   50000,  64,  32, ENC0, src0, dst0, mask0, blockout);
  pool(blockout, 50000, 40000, POOL0, 32, src0, dst0, mask0, xp1, s1, d1, m1);
  gat_block(xp1, 40000,  32,  64, ENC1, s1, d1, m1, blockout);
  pool(blockout, 40000, 24000, POOL1, 64, s1, d1, m1, xp2, s2, d2, m2);
  gat_block(xp2, 24000,  64, 128, ENC2, s2, d2, m2, blockout);
  pool(blockout, 24000,  9600, POOL2, 128, s2, d2, m2, xp3, s3, d3, m3);
  // ---------------- bottleneck
  gat_block(xp3, 9600, 128, 256, BOT, s3, d3, m3, blockout);          // [9600,256]
  // ---------------- decoder (modulo upsample + skip concat, written as f16 A)
  k_concat_f16<<<cdiv(24000L * 320, TB), TB, 0, stream>>>(blockout, 9600, 256, xp2, 64, A16, 24000);
  gat_block(nullptr, 24000, 320, 128, DEC0, s2, d2, m2, blockout);    // [24000,128]
  k_concat_f16<<<cdiv(40000L * 160, TB), TB, 0, stream>>>(blockout, 24000, 128, xp1, 32, A16, 40000);
  gat_block(nullptr, 40000, 160, 64, DEC1, s1, d1, m1, blockout);     // [40000,64]
  k_concat_f16<<<cdiv(50000L * 128, TB), TB, 0, stream>>>(blockout, 40000, 64, x, 64, A16, 50000);
  gat_block(nullptr, 50000, 128, 32, DEC2, src0, dst0, mask0, blockout); // [50000,32]
  // ---------------- output GATConv (1 head, no BN)
  gat_conv(blockout, 50000, 32, 64, 1, OUT_W, OUT_AS, OUT_AD, OUT_B,
           src0, dst0, mask0, (float*)d_out);                         // [50000,64]
}
